// MTCNN_51857435132262
// MI455X (gfx1250) — matrix operand order repaired and checked by reading, not compiled, not run on hardware
//
#include <hip/hip_runtime.h>

typedef __attribute__((ext_vector_type(16))) _Float16 v16h;
typedef __attribute__((ext_vector_type(8)))  float    v8f;

static inline int ceilDiv(int a, int b) { return (a + b - 1) / b; }

// ---------------------------------------------------------------------------
// Weight packing: f32 (OIHW conv or [O,K] dense) -> f16 in WMMA B-fragment
// order.  Element idx = ((ntile*Ksteps + kstep)*32 + lane)*16 + t maps to
// k = kstep*32 + 8*(lane>>4) + (t&7) + 16*(t>>3),  n = ntile*16 + (lane&15).
// Padded k/n entries are ZERO -- this also makes A-side padding free: garbage
// A values in padded k columns multiply with 0.0 weights.
// ---------------------------------------------------------------------------
struct PackParams {
  const float* W;
  _Float16* Bp;
  int Kreal, Kpad, Nreal, Npad;
  int IC, KH, KW;
  int isConv;
};

__global__ void pack_weights_k(PackParams p) {
  const int ksteps = p.Kpad >> 5;
  const int total = (p.Npad >> 4) * ksteps * 512;
  for (int idx = blockIdx.x * blockDim.x + threadIdx.x; idx < total;
       idx += gridDim.x * blockDim.x) {
    int t = idx & 15;
    int lane = (idx >> 4) & 31;
    int rest = idx >> 9;
    int kstep = rest % ksteps;
    int ntile = rest / ksteps;
    int k = (kstep << 5) + ((lane >> 4) << 3) + (t & 7) + ((t >> 3) << 4);
    int n = (ntile << 4) + (lane & 15);
    float v = 0.0f;
    if (k < p.Kreal && n < p.Nreal) {
      if (p.isConv) {
        int khw = p.KH * p.KW;
        int ic = k / khw;
        int r = k - ic * khw;
        int kh = r / p.KW;
        int kw = r - kh * p.KW;
        v = p.W[((n * p.IC + ic) * p.KH + kh) * p.KW + kw];
      } else {
        v = p.W[(size_t)n * p.Kreal + k];
      }
    }
    p.Bp[idx] = (_Float16)v;
  }
}

// ---------------------------------------------------------------------------
// WMMA GEMM with implicit-im2col / flatten / identity A sources.
//   AMODE 0: A = NCHW activations, rows are conv patches (implicit im2col)
//   AMODE 1: A = NCHW activations, rows flattened NWHC (torch permute(0,3,2,1))
//   AMODE 2: A = row-major f32 [M x ldA], K multiple of 32 (vector loads)
//   OMODE 0: out NCHW (conv);  OMODE 1: out row-major [M x Nreal]
// k -> source-offset map precomputed once per block into LDS; padded entries
// are clamped to offset 0 (safe load, zeroed by the zero B weights), so the
// inner loop is UNCONDITIONAL gather + WMMA: no branches, no EXEC toggling.
// Epilogue decomposes the first output row once and carry-increments the rest.
// ---------------------------------------------------------------------------
struct GemmP {
  const float* A;
  const _Float16* Bp;
  const float* bias;
  const float* alpha;   // nullptr -> no PReLU
  float* out;
  int M, Kreal, Kpad, Nreal;
  int IC, IH, IW, KH, KW, OH, OW, OC;  // AMODE 0 geometry (OC == Nreal)
  int FC, FH, FW;                      // AMODE 1 source NCHW dims
  int ldA;                             // AMODE 2
};

#define MAX_KPAD 1184

template <int AMODE>
__device__ __forceinline__ const float* a_base(const GemmP& p, int m) {
  if (AMODE == 0) {
    int ow = m % p.OW; int t = m / p.OW; int oh = t % p.OH; int b = t / p.OH;
    return p.A + (size_t)b * p.IC * p.IH * p.IW + (size_t)oh * p.IW + ow;
  } else if (AMODE == 1) {
    return p.A + (size_t)m * p.FC * p.FH * p.FW;
  } else {
    return p.A + (size_t)m * p.ldA;
  }
}

template <int AMODE>
__device__ __forceinline__ int a_off(const GemmP& p, int k) {
  if (k >= p.Kreal) return 0;   // safe clamp; zero B weights kill the product
  if (AMODE == 0) {
    int khw = p.KH * p.KW;
    int ic = k / khw;
    int r = k - ic * khw;
    int kh = r / p.KW;
    int kw = r - kh * p.KW;
    return (ic * p.IH + kh) * p.IW + kw;
  } else if (AMODE == 1) {
    int c = k % p.FC;
    int t = k / p.FC;
    int h = t % p.FH;
    int w = t / p.FH;
    return (c * p.FH + h) * p.FW + w;
  } else {
    return k;
  }
}

template <int AMODE, int OMODE>
__global__ __launch_bounds__(128) void wmma_gemm_k(GemmP p) {
  __shared__ int offTab[MAX_KPAD];

  const int lane = threadIdx.x & 31;
  const int wave = threadIdx.x >> 5;
  const int Mtiles = (p.M + 15) >> 4;
  const int mtile = blockIdx.x * 4 + wave;
  const int ntile = blockIdx.y;

  if (AMODE != 2) {
    // geometry table: div/mod chains once per block, not per element
    for (int k = threadIdx.x; k < p.Kpad; k += 128)
      offTab[k] = a_off<AMODE>(p, k);
    __syncthreads();
  }

  if (mtile >= Mtiles) return;        // wave-uniform: EXEC stays all-ones

  const int r = lane & 15;
  const int half = lane >> 4;
  int m = mtile * 16 + r;
  if (m >= p.M) m = p.M - 1;          // clamp for safe loads; masked at store
  const float* abase = a_base<AMODE>(p, m);

  const int ksteps = p.Kpad >> 5;
  const _Float16* bsrc =
      p.Bp + (((size_t)ntile * ksteps) << 9) + ((size_t)lane << 4);
  const int4* tab4 = (const int4*)offTab;

  v8f acc = {0.f, 0.f, 0.f, 0.f, 0.f, 0.f, 0.f, 0.f};

  for (int ks = 0; ks < ksteps; ++ks) {
    // A fragment: lane<16 -> K {0..7, 16..23}; lane>=16 -> K {8..15, 24..31}
    v16h a;
    if (AMODE == 2) {
      // contiguous k-runs: two aligned float4 loads per 8-wide run
      const float* src = abase + (ks << 5) + half * 8;
      float4 f0 = *(const float4*)(src);
      float4 f1 = *(const float4*)(src + 4);
      float4 f2 = *(const float4*)(src + 16);
      float4 f3 = *(const float4*)(src + 20);
      a[0] = (_Float16)f0.x;  a[1] = (_Float16)f0.y;
      a[2] = (_Float16)f0.z;  a[3] = (_Float16)f0.w;
      a[4] = (_Float16)f1.x;  a[5] = (_Float16)f1.y;
      a[6] = (_Float16)f1.z;  a[7] = (_Float16)f1.w;
      a[8] = (_Float16)f2.x;  a[9] = (_Float16)f2.y;
      a[10] = (_Float16)f2.z; a[11] = (_Float16)f2.w;
      a[12] = (_Float16)f3.x; a[13] = (_Float16)f3.y;
      a[14] = (_Float16)f3.z; a[15] = (_Float16)f3.w;
    } else {
      const int i0 = ((ks << 5) + half * 8) >> 2;  // int4 index, 32B aligned
      int offs[16];
      *(int4*)&offs[0]  = tab4[i0];
      *(int4*)&offs[4]  = tab4[i0 + 1];
      *(int4*)&offs[8]  = tab4[i0 + 4];            // +16 k == +4 int4
      *(int4*)&offs[12] = tab4[i0 + 5];
#pragma unroll
      for (int t = 0; t < 16; ++t)
        a[t] = (_Float16)abase[offs[t]];           // unconditional gather
    }

    // B fragment: pre-packed, 16 halfs per lane in A's K order (two b128 loads)
    v16h b = *(const v16h*)(bsrc + ((size_t)ks << 9));

    acc = __builtin_amdgcn_wmma_f32_16x16x32_f16(false, a, false, b,
                                                 (short)0, acc, false, false);
  }

  // Epilogue: C/D layout -> lane&15 = N column, VGPR rr -> M = rr + 8*half
  const int n = ntile * 16 + r;
  if (n >= p.Nreal) return;
  const float bia = p.bias[n];
  const bool hasA = (p.alpha != nullptr);
  const float al = hasA ? p.alpha[n] : 0.0f;

  const int mm0 = mtile * 16 + half * 8;           // 8 consecutive rows
  if (OMODE == 0) {
    // decompose once, then carry-increment (no per-row division)
    int ow = mm0 % p.OW;
    int t = mm0 / p.OW;
    int oh = t % p.OH;
    int b = t / p.OH;
#pragma unroll
    for (int rr = 0; rr < 8; ++rr) {
      if (mm0 + rr < p.M) {
        float v = acc[rr] + bia;
        if (hasA && v < 0.0f) v *= al;
        p.out[(((size_t)b * p.OC + n) * p.OH + oh) * p.OW + ow] = v;
      }
      ++ow;
      if (ow == p.OW) {                            // carry: cmp + cndmask
        ow = 0;
        ++oh;
        if (oh == p.OH) { oh = 0; ++b; }
      }
    }
  } else {
#pragma unroll
    for (int rr = 0; rr < 8; ++rr) {
      if (mm0 + rr < p.M) {
        float v = acc[rr] + bia;
        if (hasA && v < 0.0f) v *= al;
        p.out[(size_t)(mm0 + rr) * p.Nreal + n] = v;
      }
    }
  }
}

// ---------------------------------------------------------------------------
// ceil-mode maxpool (bottom/right -inf padding == clamping the window)
// ---------------------------------------------------------------------------
__global__ void maxpool_k(const float* in, float* out, int N, int C, int H,
                          int W, int OH, int OW, int k, int s) {
  size_t total = (size_t)N * C * OH * OW;
  for (size_t i = (size_t)blockIdx.x * blockDim.x + threadIdx.x; i < total;
       i += (size_t)gridDim.x * blockDim.x) {
    int ow = (int)(i % OW);
    size_t t = i / OW;
    int oh = (int)(t % OH); t /= OH;
    int c = (int)(t % C);
    int b = (int)(t / C);
    int h0 = oh * s, w0 = ow * s;
    int h1 = h0 + k; if (h1 > H) h1 = H;
    int w1 = w0 + k; if (w1 > W) w1 = W;
    const float* src = in + ((size_t)b * C + c) * H * W;
    float mx = -3.402823466e38f;
    for (int h = h0; h < h1; ++h)
      for (int w = w0; w < w1; ++w)
        mx = fmaxf(mx, src[(size_t)h * W + w]);
    out[i] = mx;
  }
}

// two-way softmax along a strided channel pair (NCHW C=2 or [M,2] rows)
__global__ void softmax2_k(const float* in, float* out, int outer, int inner) {
  size_t total = (size_t)outer * inner;
  for (size_t i = (size_t)blockIdx.x * blockDim.x + threadIdx.x; i < total;
       i += (size_t)gridDim.x * blockDim.x) {
    int o = (int)(i / inner);
    int r = (int)(i % inner);
    size_t i0 = (size_t)o * 2 * inner + r;
    size_t i1 = i0 + inner;
    float x0 = in[i0], x1 = in[i1];
    float mx = fmaxf(x0, x1);
    float e0 = __expf(x0 - mx), e1 = __expf(x1 - mx);
    float inv = 1.0f / (e0 + e1);
    out[i0] = e0 * inv;
    out[i1] = e1 * inv;
  }
}

// ---------------------------------------------------------------------------
extern "C" void kernel_launch(void* const* d_in, const int* in_sizes, int n_in,
                              void* d_out, int out_size, void* d_ws,
                              size_t ws_size, hipStream_t stream) {
  (void)in_sizes; (void)n_in; (void)out_size; (void)ws_size;
  const float* img   = (const float*)d_in[0];
  const float* rcrop = (const float*)d_in[1];
  const float* ocrop = (const float*)d_in[2];
  auto P = [&](int i) { return (const float*)d_in[i]; };

  // bump allocator in d_ws
  size_t off = 0;
  auto wsAlloc = [&](size_t bytes) -> char* {
    off = (off + 255) & ~(size_t)255;
    char* p = (char*)d_ws + off;
    off += bytes;
    return p;
  };
  _Float16* wpack = (_Float16*)wsAlloc((size_t)1 << 20);  // reused per layer
  const size_t ARENA_A = 277348352;  // onet conv1 out (largest even-parity buf)
  const size_t ARENA_B = 69337088;   // onet pool1 out (largest odd-parity buf)
  float* bufA = (float*)wsAlloc(ARENA_A * sizeof(float));
  float* bufB = (float*)wsAlloc(ARENA_B * sizeof(float));

  auto packConv = [&](const float* W, int OC, int IC, int KH, int KW) {
    PackParams pp{};
    pp.W = W; pp.Bp = wpack;
    pp.Kreal = IC * KH * KW; pp.Kpad = (pp.Kreal + 31) & ~31;
    pp.Nreal = OC; pp.Npad = (OC + 15) & ~15;
    pp.IC = IC; pp.KH = KH; pp.KW = KW; pp.isConv = 1;
    int total = (pp.Npad >> 4) * (pp.Kpad >> 5) * 512;
    pack_weights_k<<<ceilDiv(total, 256), 256, 0, stream>>>(pp);
  };
  auto packDense = [&](const float* W, int N, int K) {
    PackParams pp{};
    pp.W = W; pp.Bp = wpack;
    pp.Kreal = K; pp.Kpad = (K + 31) & ~31;
    pp.Nreal = N; pp.Npad = (N + 15) & ~15;
    pp.isConv = 0;
    int total = (pp.Npad >> 4) * (pp.Kpad >> 5) * 512;
    pack_weights_k<<<ceilDiv(total, 256), 256, 0, stream>>>(pp);
  };

  auto conv = [&](const float* A, int B_, int IC, int IH, int IW, int KH,
                  int KW, int OC, const float* W, const float* bias,
                  const float* alpha, float* out) {
    packConv(W, OC, IC, KH, KW);
    GemmP g{};
    int OH = IH - KH + 1, OW = IW - KW + 1;
    g.A = A; g.Bp = wpack; g.bias = bias; g.alpha = alpha; g.out = out;
    g.M = B_ * OH * OW; g.Kreal = IC * KH * KW; g.Kpad = (g.Kreal + 31) & ~31;
    g.Nreal = OC; g.IC = IC; g.IH = IH; g.IW = IW; g.KH = KH; g.KW = KW;
    g.OH = OH; g.OW = OW; g.OC = OC;
    dim3 grid(ceilDiv(ceilDiv(g.M, 16), 4), (OC + 15) >> 4);
    wmma_gemm_k<0, 0><<<grid, 128, 0, stream>>>(g);
  };

  auto denseFlat = [&](const float* A, int B_, int FC, int FH, int FW, int N,
                       const float* W, const float* bias, const float* alpha,
                       float* out) {
    int K = FC * FH * FW;
    packDense(W, N, K);
    GemmP g{};
    g.A = A; g.Bp = wpack; g.bias = bias; g.alpha = alpha; g.out = out;
    g.M = B_; g.Kreal = K; g.Kpad = (K + 31) & ~31; g.Nreal = N;
    g.FC = FC; g.FH = FH; g.FW = FW;
    dim3 grid(ceilDiv(ceilDiv(B_, 16), 4), (N + 15) >> 4);
    wmma_gemm_k<1, 1><<<grid, 128, 0, stream>>>(g);
  };

  auto denseRow = [&](const float* A, int B_, int K, int N, const float* W,
                      const float* bias, const float* alpha, float* out) {
    packDense(W, N, K);
    GemmP g{};
    g.A = A; g.Bp = wpack; g.bias = bias; g.alpha = alpha; g.out = out;
    g.M = B_; g.Kreal = K; g.Kpad = (K + 31) & ~31; g.Nreal = N; g.ldA = K;
    dim3 grid(ceilDiv(ceilDiv(B_, 16), 4), (N + 15) >> 4);
    wmma_gemm_k<2, 1><<<grid, 128, 0, stream>>>(g);
  };

  auto maxpool = [&](const float* in, float* out, int N, int C, int H, int W,
                     int k, int s) {
    int OH = ceilDiv(H - k, s) + 1;
    int OW = ceilDiv(W - k, s) + 1;
    size_t total = (size_t)N * C * OH * OW;
    size_t blk = (total + 255) / 256;
    if (blk > (size_t)(1 << 20)) blk = (size_t)(1 << 20);
    maxpool_k<<<dim3((unsigned)blk), 256, 0, stream>>>(in, out, N, C, H, W, OH,
                                                       OW, k, s);
  };

  auto softmax2 = [&](const float* in, float* out, int outer, int inner) {
    size_t total = (size_t)outer * inner;
    size_t blk = (total + 255) / 256;
    if (blk > (size_t)(1 << 20)) blk = (size_t)(1 << 20);
    softmax2_k<<<dim3((unsigned)blk), 256, 0, stream>>>(in, out, outer, inner);
  };

  // output layout: (pb, pa, rb, ra, ob, oc, oa) concatenated flat
  float* outF = (float*)d_out;
  const size_t PB  = 0;
  const size_t PA  = PB  + (size_t)4 * 4 * 355 * 635;
  const size_t RB  = PA  + (size_t)4 * 2 * 355 * 635;
  const size_t RA  = RB  + (size_t)8192 * 4;
  const size_t OB  = RA  + (size_t)8192 * 2;
  const size_t OCo = OB  + (size_t)4096 * 4;
  const size_t OA  = OCo + (size_t)4096 * 10;

  // ---- PNet: img (4,3,720,1280) ----
  conv(img, 4, 3, 720, 1280, 3, 3, 10, P(3), P(4), P(5), bufA);       // 718x1278
  maxpool(bufA, bufB, 4, 10, 718, 1278, 2, 2);                        // 359x639
  conv(bufB, 4, 10, 359, 639, 3, 3, 16, P(6), P(7), P(8), bufA);      // 357x637
  conv(bufA, 4, 16, 357, 637, 3, 3, 32, P(9), P(10), P(11), bufB);    // 355x635
  conv(bufB, 4, 32, 355, 635, 1, 1, 2, P(12), P(13), nullptr, bufA);  // cls logits
  softmax2(bufA, outF + PA, 4, 355 * 635);                            // pa
  conv(bufB, 4, 32, 355, 635, 1, 1, 4, P(14), P(15), nullptr, outF + PB); // pb

  // ---- RNet: crops (8192,3,24,24) ----
  conv(rcrop, 8192, 3, 24, 24, 3, 3, 28, P(16), P(17), P(18), bufA);  // 22x22
  maxpool(bufA, bufB, 8192, 28, 22, 22, 3, 2);                        // 11x11
  conv(bufB, 8192, 28, 11, 11, 3, 3, 48, P(19), P(20), P(21), bufA);  // 9x9
  maxpool(bufA, bufB, 8192, 48, 9, 9, 3, 2);                          // 4x4
  conv(bufB, 8192, 48, 4, 4, 2, 2, 64, P(22), P(23), P(24), bufA);    // 3x3
  denseFlat(bufA, 8192, 64, 3, 3, 128, P(25), P(26), P(27), bufB);    // [8192,128]
  denseRow(bufB, 8192, 128, 2, P(28), P(29), nullptr, bufA);          // cls logits
  softmax2(bufA, outF + RA, 8192, 1);                                 // ra
  denseRow(bufB, 8192, 128, 4, P(30), P(31), nullptr, outF + RB);     // rb

  // ---- ONet: crops (4096,3,48,48) ----
  conv(ocrop, 4096, 3, 48, 48, 3, 3, 32, P(32), P(33), P(34), bufA);  // 46x46
  maxpool(bufA, bufB, 4096, 32, 46, 46, 3, 2);                        // 23x23
  conv(bufB, 4096, 32, 23, 23, 3, 3, 64, P(35), P(36), P(37), bufA);  // 21x21
  maxpool(bufA, bufB, 4096, 64, 21, 21, 3, 2);                        // 10x10
  conv(bufB, 4096, 64, 10, 10, 3, 3, 64, P(38), P(39), P(40), bufA);  // 8x8
  maxpool(bufA, bufB, 4096, 64, 8, 8, 2, 2);                          // 4x4
  conv(bufB, 4096, 64, 4, 4, 2, 2, 128, P(41), P(42), P(43), bufA);   // 3x3
  denseFlat(bufA, 4096, 128, 3, 3, 256, P(44), P(45), P(46), bufB);   // [4096,256]
  denseRow(bufB, 4096, 256, 2, P(47), P(48), nullptr, bufA);          // cls logits
  softmax2(bufA, outF + OA, 4096, 1);                                 // oa
  denseRow(bufB, 4096, 256, 4, P(49), P(50), nullptr, outF + OB);     // ob
  denseRow(bufB, 4096, 256, 10, P(51), P(52), nullptr, outF + OCo);   // oc
}
